// FRAP_50457275793809
// MI455X (gfx1250) — matrix operand order repaired and checked by reading, not compiled, not run on hardware
//
#include <hip/hip_runtime.h>
#include <hip/hip_bf16.h>
#include <hip/hip_fp16.h>

typedef __attribute__((ext_vector_type(16))) _Float16 v16h;
typedef __attribute__((ext_vector_type(8)))  float    v8f;

#define NO      8     // O
#define MLANES  12    // M
#define LE      16
#define NPAIRS  56    // O*(O-1)
#define WAVES   4
#define ELEMS   16    // batch elements per wave (WMMA M dim)

#define REL_CNT   (NPAIRS * 20)          // floats in ws
#define PEMB_CNT  8                      // sigmoid(p_emb), floats in ws
#define FRAG_CNT  (4 * 512)              // f16 fragments: lconv0,lconv1,hid0,hid1

#if defined(__HIP_DEVICE_COMPILE__) && __has_builtin(__builtin_amdgcn_global_load_async_to_lds_b32)
#define HAVE_ASYNC_LDS 1
#else
#define HAVE_ASYNC_LDS 0
#endif

__device__ __forceinline__ float sigmoidf_(float x) { return 1.0f / (1.0f + __expf(-x)); }
// single v_max_num_f32 (fmax has NaN-quieting semantics built in)
__device__ __forceinline__ float reluf_(float x)    { return __builtin_fmaxf(x, 0.0f); }

// XOR-swizzle add within 16-lane groups (wave32): and=0x1F, or=0, xor=m
#define SWZ_ADD(s, m)                                                          \
    s += __int_as_float(__builtin_amdgcn_ds_swizzle(__float_as_int(s),         \
                                                    ((m) << 10) | 0x1F))

// ---------------------------------------------------------------------------
// Prep kernel: batch-independent work.
//  - rel[p][n] = relu(relu(rel_emb[comp_mask[p]]) @ rconv_W + rconv_b) (56x20)
//  - sigmoid(p_emb) (8 values)
//  - f16 B-matrix fragments (CDNA5 16-bit B layout: element i of lane L holds
//    K = (i&7) + 8*(L>=16) + 16*(i>=8), N = (L&15) + 16*tile_parity) for lconv_W and hid_W.
// ---------------------------------------------------------------------------
__global__ void frap_prep(const int* __restrict__ comp_mask,
                          const float* __restrict__ rel_emb,
                          const float* __restrict__ rconv_W,
                          const float* __restrict__ rconv_b,
                          const float* __restrict__ lconv_W,
                          const float* __restrict__ hid_W,
                          const float* __restrict__ p_emb,
                          float* __restrict__ ws_rel,
                          float* __restrict__ ws_pemb,
                          _Float16* __restrict__ ws_frag)
{
    int t = threadIdx.x;
    for (int i = t; i < REL_CNT; i += blockDim.x) {
        int p = i / 20, n = i % 20;
        int cm = comp_mask[p];
        float acc = rconv_b[n];
        #pragma unroll
        for (int k = 0; k < 4; ++k) {
            float rv = reluf_(rel_emb[cm * 4 + k]);
            acc += rv * rconv_W[k * 20 + n];
        }
        ws_rel[i] = reluf_(acc);
    }
    if (t < PEMB_CNT) ws_pemb[t] = sigmoidf_(p_emb[t]);
    for (int i = t; i < FRAG_CNT; i += blockDim.x) {
        int tile = i >> 9;            // 0..3
        int lane = (i >> 4) & 31;
        int el   = i & 15;
        int K = (el & 7) + ((lane >> 4) ? 8 : 0) + ((el >> 3) ? 16 : 0);
        int N = (lane & 15) + 16 * (tile & 1);
        float v = 0.0f;
        if (tile < 2) { if (N < 20)            v = lconv_W[K * 20 + N]; }
        else          { if (N < 20 && K < 20)  v = hid_W[K * 20 + N]; }
        ws_frag[i] = (_Float16)v;
    }
}

// ---------------------------------------------------------------------------
// Main kernel: one wave per 16 batch elements. 224 WMMAs per wave-tile.
// ---------------------------------------------------------------------------
__global__ void __launch_bounds__(WAVES * 32)
frap_main(const float* __restrict__ states,
          const int*   __restrict__ phase_pairs,
          const float* __restrict__ d_W,
          const float* __restrict__ d_b,
          const float* __restrict__ lane_W,
          const float* __restrict__ lane_b,
          const float* __restrict__ lconv_b,
          const float* __restrict__ hid_b,
          const float* __restrict__ bm_W,
          const float* __restrict__ bm_b,
          const float* __restrict__ ws_rel,
          const float* __restrict__ ws_pemb,
          const _Float16* __restrict__ ws_frag,
          float* __restrict__ out, int B)
{
    __shared__ _Float16 pde_lds[WAVES][ELEMS * MLANES * LE]; // 24 KB
    __shared__ _Float16 pairs_lds[WAVES][ELEMS * NO * LE];   // 16 KB
    __shared__ _Float16 comb_lds[WAVES][ELEMS * 32];         // 4 KB
    __shared__ float    rel_s[REL_CNT];                      // 4.375 KB
    __shared__ int      pp_s[16];

    const int tid  = threadIdx.x;
    const int wave = tid >> 5;
    const int lane = tid & 31;
    const int hi   = lane >> 4;          // 0 or 1
    const int l15  = lane & 15;
    const int koff = hi ? 8 : 0;
    const int base = (blockIdx.x * WAVES + wave) * ELEMS;

    // ---- block-wide staging ----
#if HAVE_ASYNC_LDS
    for (int i = tid; i < REL_CNT; i += blockDim.x) {
        __builtin_amdgcn_global_load_async_to_lds_b32(
            (__attribute__((address_space(1))) int*)(ws_rel + i),
            (__attribute__((address_space(3))) int*)(rel_s + i),
            /*offset=*/0, /*cpol=*/0);
    }
    asm volatile("s_wait_asynccnt 0" ::: "memory");
#else
    for (int i = tid; i < REL_CNT; i += blockDim.x) rel_s[i] = ws_rel[i];
#endif
    if (tid < 16) pp_s[tid] = phase_pairs[tid];
    // zero comb buffer once: columns 20..31 must stay zero (hid K-padding)
    for (int i = tid; i < WAVES * ELEMS * 32; i += blockDim.x)
        ((_Float16*)comb_lds)[i] = (_Float16)0.0f;

    // ---- per-lane B fragments (stay in VGPRs for all 56 pair iterations) ----
    const v16h* fp = (const v16h*)ws_frag;   // 32 bytes per lane per tile
    v16h Bl0 = fp[0 * 32 + lane];
    v16h Bl1 = fp[1 * 32 + lane];
    v16h Bh0 = fp[2 * 32 + lane];
    v16h Bh1 = fp[3 * 32 + lane];

    float pemb_sig[8];
    #pragma unroll
    for (int k = 0; k < 8; ++k) pemb_sig[k] = ws_pemb[k];

    __syncthreads();

    if (base + ELEMS > B) return;

    // ---- pde: 192 rows per wave (16 elems x 12 lanes), VALU (K=8 is tiny) ----
    #pragma unroll
    for (int it = 0; it < (ELEMS * MLANES) / 32; ++it) {
        int idx = it * 32 + lane;         // 0..191
        int e   = idx / MLANES;
        int row = idx % MLANES;
        const float* st = states + (size_t)(base + e) * 13;
        int act = (int)st[0];
        int ppa = pp_s[2 * act], ppb = pp_s[2 * act + 1];
        int ext = (row == ppa || row == ppb) ? 1 : 0;
        float pd[8];
        #pragma unroll
        for (int k = 0; k < 4; ++k) pd[k] = pemb_sig[ext * 4 + k];
        float dm = st[1 + row];           // DS==1: gi[row,0] == row
        #pragma unroll
        for (int k = 0; k < 4; ++k) pd[4 + k] = sigmoidf_(dm * d_W[k] + d_b[k]);
        _Float16* dst = &pde_lds[wave][idx * LE];
        #pragma unroll
        for (int n = 0; n < LE; ++n) {
            float acc = lane_b[n];
            #pragma unroll
            for (int k = 0; k < 8; ++k) acc += pd[k] * lane_W[k * LE + n];
            dst[n] = (_Float16)reluf_(acc);
        }
    }

    // ---- pairs[e][r][:] = pde[e][pp[r,0]][:] + pde[e][pp[r,1]][:], f16 ----
    #pragma unroll
    for (int it = 0; it < (ELEMS * NO) / 32; ++it) {   // 4 iterations
        int idx = it * 32 + lane;       // 0..127
        int e = idx >> 3, r = idx & 7;
        int p0 = pp_s[2 * r], p1 = pp_s[2 * r + 1];
        const _Float16* s0 = &pde_lds[wave][(e * MLANES + p0) * LE];
        const _Float16* s1 = &pde_lds[wave][(e * MLANES + p1) * LE];
        _Float16* d = &pairs_lds[wave][idx * LE];
        #pragma unroll
        for (int n = 0; n < LE; ++n) d[n] = s0[n] + s1[n];
    }
    asm volatile("s_wait_dscnt 0" ::: "memory");

    // ---- per-lane constants for C-layout columns (lane = N) ----
    const float bmb  = bm_b[0];
    const float bmw0 = bm_W[l15];
    const float bmw1 = (l15 < 4) ? bm_W[16 + l15] : 0.0f;
    const float lcb0 = lconv_b[l15];
    const float lcb1 = (l15 < 4) ? lconv_b[16 + l15] : 0.0f;
    const float hdb0 = hid_b[l15];
    const float hdb1 = (l15 < 4) ? hid_b[16 + l15] : 0.0f;

    const _Float16* prow = &pairs_lds[wave][l15 * NO * LE]; // A rows: M = l15
    _Float16* cl = &comb_lds[wave][0];

    for (int o = 0; o < NO; ++o) {
        float qacc[8];
        #pragma unroll
        for (int r = 0; r < 8; ++r) qacc[r] = 0.0f;

        // K 0..15 half of A depends only on o — load once
        _Float16 alo[8];
        #pragma unroll
        for (int k = 0; k < 8; ++k) alo[k] = prow[o * LE + koff + k];

        for (int jj = 0; jj < 7; ++jj) {
            const int j = jj + (jj >= o ? 1 : 0);
            const int p = o * 7 + jj;

            // A fragment (16x32 f16): K 0..15 = pairs[o], K 16..31 = pairs[j]
            v16h A;
            #pragma unroll
            for (int k = 0; k < 8; ++k) A[k] = alo[k];
            const _Float16* pj = &prow[j * LE + koff];
            #pragma unroll
            for (int k = 0; k < 8; ++k) A[8 + k] = pj[k];

            // lconv: two N-tiles, bias preloaded into C
            v8f c0, c1;
            #pragma unroll
            for (int r = 0; r < 8; ++r) { c0[r] = lcb0; c1[r] = lcb1; }
            c0 = __builtin_amdgcn_wmma_f32_16x16x32_f16(false, A, false, Bl0,
                                                        (short)0, c0, false, false);
            c1 = __builtin_amdgcn_wmma_f32_16x16x32_f16(false, A, false, Bl1,
                                                        (short)0, c1, false, false);

            const float rel0 = rel_s[p * 20 + l15];
            const float rel1 = (l15 < 4) ? rel_s[p * 20 + 16 + l15] : 0.0f;

            // relu * rel, re-layout C(lane=N) -> A(lane=M) through LDS
            #pragma unroll
            for (int r = 0; r < 8; ++r) {
                int m = r + (hi ? 8 : 0);
                cl[m * 32 + l15] = (_Float16)(reluf_(c0[r]) * rel0);
                if (l15 < 4)
                    cl[m * 32 + 16 + l15] = (_Float16)(reluf_(c1[r]) * rel1);
            }
            asm volatile("s_wait_dscnt 0" ::: "memory");

            v16h A2;
            {
                const _Float16* cm = &comb_lds[wave][l15 * 32];
                #pragma unroll
                for (int k = 0; k < 8; ++k) A2[k] = cm[koff + k];
                #pragma unroll
                for (int k = 0; k < 8; ++k) A2[8 + k] = cm[16 + koff + k];
            }

            // hid: two N-tiles (K padded to 32 with zeroed cols 20..31)
            v8f d0, d1;
            #pragma unroll
            for (int r = 0; r < 8; ++r) { d0[r] = hdb0; d1[r] = hdb1; }
            d0 = __builtin_amdgcn_wmma_f32_16x16x32_f16(false, A2, false, Bh0,
                                                        (short)0, d0, false, false);
            d1 = __builtin_amdgcn_wmma_f32_16x16x32_f16(false, A2, false, Bh1,
                                                        (short)0, d1, false, false);

            // bm dot partial, kept per-lane; reduced once per o below
            #pragma unroll
            for (int r = 0; r < 8; ++r)
                qacc[r] += reluf_(d0[r]) * bmw0 + reluf_(d1[r]) * bmw1;
        }

        // cross-lane sum over N within each 16-lane group (once per o)
        #pragma unroll
        for (int r = 0; r < 8; ++r) {
            float s = qacc[r];
            SWZ_ADD(s, 1);
            SWZ_ADD(s, 2);
            SWZ_ADD(s, 4);
            SWZ_ADD(s, 8);
            qacc[r] = s + 7.0f * bmb;
        }
        // lanes 0 / 16 own rows 0..7 / 8..15 after the group reduction
        if (l15 == 0) {
            #pragma unroll
            for (int r = 0; r < 8; ++r) {
                int m = r + (hi ? 8 : 0);
                out[(size_t)(base + m) * NO + o] = qacc[r];
            }
        }
    }
}

// ---------------------------------------------------------------------------
extern "C" void kernel_launch(void* const* d_in, const int* in_sizes, int n_in,
                              void* d_out, int out_size, void* d_ws, size_t ws_size,
                              hipStream_t stream)
{
    const float* states      = (const float*)d_in[0];
    const int*   phase_pairs = (const int*)  d_in[1];
    const int*   comp_mask   = (const int*)  d_in[2];
    const float* p_emb       = (const float*)d_in[3];
    const float* d_W         = (const float*)d_in[4];
    const float* d_b         = (const float*)d_in[5];
    const float* lane_W      = (const float*)d_in[6];
    const float* lane_b      = (const float*)d_in[7];
    const float* lconv_W     = (const float*)d_in[8];
    const float* lconv_b     = (const float*)d_in[9];
    const float* rel_emb     = (const float*)d_in[10];
    const float* rconv_W     = (const float*)d_in[11];
    const float* rconv_b     = (const float*)d_in[12];
    const float* hid_W       = (const float*)d_in[13];
    const float* hid_b       = (const float*)d_in[14];
    const float* bm_W        = (const float*)d_in[15];
    const float* bm_b        = (const float*)d_in[16];

    const int B = in_sizes[0] / 13;

    float*    ws_rel  = (float*)d_ws;
    float*    ws_pemb = ws_rel + REL_CNT;
    _Float16* ws_frag = (_Float16*)(ws_pemb + PEMB_CNT);

    frap_prep<<<1, 256, 0, stream>>>(comp_mask, rel_emb, rconv_W, rconv_b,
                                     lconv_W, hid_W, p_emb,
                                     ws_rel, ws_pemb, ws_frag);

    const int per_block = WAVES * ELEMS;
    const int blocks = (B + per_block - 1) / per_block;
    frap_main<<<blocks, WAVES * 32, 0, stream>>>(
        states, phase_pairs, d_W, d_b, lane_W, lane_b,
        lconv_b, hid_b, bm_W, bm_b, ws_rel, ws_pemb, ws_frag,
        (float*)d_out, B);
}
